// Classifier_1168231104747
// MI455X (gfx1250) — compile-verified, matrix operand order checked by reading, not run on hardware
//
#include <hip/hip_runtime.h>
#include <hip/hip_bf16.h>

// ---------------------------------------------------------------------------
// Shapes (from reference)
// ---------------------------------------------------------------------------
#define S0c   2
#define S1c   64
#define Tc    25
#define Bc    3200          // S0*S1*T images through the CNN
#define Nc    20000
#define Ec    160000
#define NTc   40000         // S0*N nodes
#define REALc 320000        // S0*E real edges
#define ETc   360000        // + S0*N self loops
#define NODEF 301
#define KP0   320           // NODEF padded to mult of 64
#define GRUH  12

typedef __bf16 v16bf __attribute__((ext_vector_type(16)));
typedef float  v8f   __attribute__((ext_vector_type(8)));
typedef unsigned int u32x4 __attribute__((ext_vector_type(4)));

union FragBF { u32x4 q[2]; v16bf v; };

__device__ __forceinline__ unsigned short f2bf(float f) {
  unsigned u = __float_as_uint(f);
  u += 0x7FFFu + ((u >> 16) & 1u);           // round-to-nearest-even
  return (unsigned short)(u >> 16);
}
__device__ __forceinline__ unsigned f2ord(float f) {
  unsigned u = __float_as_uint(f);
  return (u & 0x80000000u) ? ~u : (u | 0x80000000u);
}
__device__ __forceinline__ float ord2f(unsigned u) {
  return __uint_as_float((u & 0x80000000u) ? (u & 0x7fffffffu) : ~u);
}

// ---------------------------------------------------------------------------
// Weight preparation
// ---------------------------------------------------------------------------
// (N,K) f32 -> (N,Kp) bf16, zero-padded (ci-major conv1 weights)
__global__ void k_wpad(const float* __restrict__ w, unsigned short* __restrict__ o,
                       int N, int K, int Kp) {
  int i = blockIdx.x * blockDim.x + threadIdx.x;
  if (i >= N * Kp) return;
  int n = i / Kp, k = i % Kp;
  o[i] = (k < K) ? f2bf(w[(size_t)n * K + k]) : (unsigned short)0;
}
// conv weights (Cout, Cin, 27) -> tap-major (Cout, tap*Cin + ci) bf16
__global__ void k_wpad_tap(const float* __restrict__ w, unsigned short* __restrict__ o,
                           int Cout, int Cin) {
  int Kp = Cin * 27;
  int i = blockIdx.x * blockDim.x + threadIdx.x;
  if (i >= Cout * Kp) return;
  int co = i / Kp, r = i % Kp;
  int tap = r / Cin, ci = r % Cin;
  o[i] = f2bf(w[(size_t)co * Kp + (size_t)ci * 27 + tap]);
}
// (K,N) f32 -> transposed (N,Kp) bf16, zero-padded
__global__ void k_wtrans(const float* __restrict__ w, unsigned short* __restrict__ o,
                         int N, int K, int Kp) {
  int i = blockIdx.x * blockDim.x + threadIdx.x;
  if (i >= N * Kp) return;
  int n = i / Kp, k = i % Kp;
  o[i] = (k < K) ? f2bf(w[(size_t)k * N + n]) : (unsigned short)0;
}

// ---------------------------------------------------------------------------
// conv1 im2col offset table: T[p][k] = input element offset or -1.
// conv1: Cin=5, D=5, H=6, W=6 (P=180, K=135, Kp=160); 192 rows cover tiles.
// ---------------------------------------------------------------------------
__global__ void k_tab1(int* __restrict__ tab) {
  int i = blockIdx.x * blockDim.x + threadIdx.x;
  if (i >= 192 * 160) return;
  int p = i / 160, k = i % 160;
  int off = -1;
  if (p < 180 && k < 135) {
    int z = p / 36, y = (p % 36) / 6, x = p % 6;
    int ci = k / 27, r27 = k % 27;
    int iz = z + r27 / 9 - 1, iy = y + (r27 % 9) / 3 - 1, ix = x + r27 % 3 - 1;
    if (iz >= 0 && iz < 5 && iy >= 0 && iy < 6 && ix >= 0 && ix < 6)
      off = (ci * 5 + iz) * 36 + iy * 6 + ix;
  }
  tab[i] = off;
}

// ---------------------------------------------------------------------------
// conv1 via offset table: no div/mod in the loop, dual accumulators.
// grid = (12, 1, B), block = 128 (4 waves cover Cout=64).
// ---------------------------------------------------------------------------
__global__ __launch_bounds__(128)
void k_conv1_tab(const float* __restrict__ in, const unsigned short* __restrict__ Wt,
                 float* __restrict__ out, const float* __restrict__ bias,
                 const float* __restrict__ bg, const float* __restrict__ bb,
                 const float* __restrict__ bm, const float* __restrict__ bv,
                 const int* __restrict__ tab) {
  const int P = 180, Kp = 160, Cout = 64;
  __shared__ __align__(16) unsigned short As[16][40];
  const int tid  = threadIdx.x;
  const int wave = tid >> 5, lane = tid & 31;
  const int b  = blockIdx.z;
  const int m0 = blockIdx.x * 16;
  const int n0 = wave * 16;
  const int row = tid >> 3, seg = tid & 7;
  const int p = m0 + row;
  const int4* t4 = (const int4*)(tab + p * Kp);
  const float* inb = in + (size_t)b * 900;       // 5*180
  v8f c0 = {0.f, 0.f, 0.f, 0.f, 0.f, 0.f, 0.f, 0.f};
  v8f c1 = {0.f, 0.f, 0.f, 0.f, 0.f, 0.f, 0.f, 0.f};
  const int bk = (lane >= 16) ? 8 : 0;
  const unsigned short* wr = Wt + (size_t)(n0 + (lane & 15)) * Kp;
#pragma unroll
  for (int it = 0; it < 5; ++it) {
    const int k0 = it * 32;
    int4 tv = t4[(k0 >> 2) + seg];
    As[row][seg * 4 + 0] = (tv.x >= 0) ? f2bf(inb[tv.x]) : (unsigned short)0;
    As[row][seg * 4 + 1] = (tv.y >= 0) ? f2bf(inb[tv.y]) : (unsigned short)0;
    As[row][seg * 4 + 2] = (tv.z >= 0) ? f2bf(inb[tv.z]) : (unsigned short)0;
    As[row][seg * 4 + 3] = (tv.w >= 0) ? f2bf(inb[tv.w]) : (unsigned short)0;
    __syncthreads();
    FragBF a, bfr;
    a.q[0]   = *(const u32x4*)&As[lane & 15][bk];
    a.q[1]   = *(const u32x4*)&As[lane & 15][16 + bk];
    bfr.q[0] = *(const u32x4*)(wr + k0 + bk);
    bfr.q[1] = *(const u32x4*)(wr + k0 + bk + 16);
    if (it & 1)
      c1 = __builtin_amdgcn_wmma_f32_16x16x32_bf16(false, a.v, false, bfr.v,
                                                   (short)0, c1, false, false);
    else
      c0 = __builtin_amdgcn_wmma_f32_16x16x32_bf16(false, a.v, false, bfr.v,
                                                   (short)0, c0, false, false);
    __syncthreads();
  }
  const int n = n0 + (lane & 15);
  const float bs = bias[n];
  const float sc = bg[n] * rsqrtf(bv[n] + 1e-5f);
  const float sh = bb[n] - bm[n] * sc;
  float* ob = out + ((size_t)b * Cout + n) * P;
#pragma unroll
  for (int r = 0; r < 8; ++r) {
    int m = m0 + r + ((lane >> 4) << 3);
    if (m < P) {
      float vv = (c0[r] + c1[r] + bs) * sc + sh;
      ob[m] = vv > 0.f ? vv : 0.f;
    }
  }
}

// ---------------------------------------------------------------------------
// Tap-major implicit-GEMM conv, Cin == 64 (conv2, conv3).
// Per 3x3x3 tap: spatial decode hoisted, coalesced 16-row gather of 64
// channels into LDS, 2 independent-accumulator WMMAs (no hazard NOPs).
// grid = (ceil(P/16), Cout/64, B), block = 128 (4 waves).
// ---------------------------------------------------------------------------
__global__ __launch_bounds__(128)
void k_conv_tap(const float* __restrict__ in, const unsigned short* __restrict__ Wt,
                float* __restrict__ out, const float* __restrict__ bias,
                const float* __restrict__ bg, const float* __restrict__ bb,
                const float* __restrict__ bm, const float* __restrict__ bv,
                int D, int H, int W, int Cout) {
  const int Cin = 64;
  const int P  = D * H * W;
  const int Kp = Cin * 27;
  const int HW = H * W;
  __shared__ __align__(16) unsigned short As[16][72];
  const int tid  = threadIdx.x;
  const int wave = tid >> 5, lane = tid & 31;
  const int b  = blockIdx.z;
  const int m0 = blockIdx.x * 16;
  const int n0 = blockIdx.y * 64 + wave * 16;
  const int grow = tid & 15;            // im2col row within tile
  const int cb   = (tid >> 4) << 3;     // 8 input channels per thread
  const int p = m0 + grow;
  int z = 0, y = 0, x = 0;
  const bool pv = (p < P);
  if (pv) { z = p / HW; y = (p % HW) / W; x = p % W; }
  const float* inb = in + (size_t)b * Cin * P + (size_t)cb * P;
  v8f c0 = {0.f, 0.f, 0.f, 0.f, 0.f, 0.f, 0.f, 0.f};
  v8f c1 = {0.f, 0.f, 0.f, 0.f, 0.f, 0.f, 0.f, 0.f};
  const int bk = (lane >= 16) ? 8 : 0;
  const unsigned short* wr = Wt + (size_t)(n0 + (lane & 15)) * Kp;
  for (int tap = 0; tap < 27; ++tap) {
    const int iz = z + tap / 9 - 1;
    const int iy = y + (tap % 9) / 3 - 1;
    const int ix = x + tap % 3 - 1;
    const bool ok = pv && iz >= 0 && iz < D && iy >= 0 && iy < H && ix >= 0 && ix < W;
    const float* src = inb + ((size_t)iz * HW + (size_t)iy * W + ix);
    if (tap < 26) __builtin_prefetch(wr + (tap + 1) * Cin, 0, 1);
    union { unsigned short s[8]; u32x4 q; } tmp;
#pragma unroll
    for (int j = 0; j < 8; ++j)
      tmp.s[j] = ok ? f2bf(src[(size_t)j * P]) : (unsigned short)0;
    *(u32x4*)&As[grow][cb] = tmp.q;
    __syncthreads();
    FragBF a0, a1, b0, b1;
    a0.q[0] = *(const u32x4*)&As[lane & 15][bk];
    a0.q[1] = *(const u32x4*)&As[lane & 15][16 + bk];
    a1.q[0] = *(const u32x4*)&As[lane & 15][32 + bk];
    a1.q[1] = *(const u32x4*)&As[lane & 15][48 + bk];
    const unsigned short* wt = wr + tap * Cin;
    b0.q[0] = *(const u32x4*)(wt + bk);
    b0.q[1] = *(const u32x4*)(wt + 16 + bk);
    b1.q[0] = *(const u32x4*)(wt + 32 + bk);
    b1.q[1] = *(const u32x4*)(wt + 48 + bk);
    c0 = __builtin_amdgcn_wmma_f32_16x16x32_bf16(false, a0.v, false, b0.v,
                                                 (short)0, c0, false, false);
    c1 = __builtin_amdgcn_wmma_f32_16x16x32_bf16(false, a1.v, false, b1.v,
                                                 (short)0, c1, false, false);
    __syncthreads();
  }
  const int n = n0 + (lane & 15);
  const float bs = bias[n];
  const float sc = bg[n] * rsqrtf(bv[n] + 1e-5f);
  const float sh = bb[n] - bm[n] * sc;
  float* ob = out + ((size_t)b * Cout + n) * P;
#pragma unroll
  for (int r = 0; r < 8; ++r) {
    int m = m0 + r + ((lane >> 4) << 3);
    if (m < P) {
      float vv = (c0[r] + c1[r] + bs) * sc + sh;
      ob[m] = vv > 0.f ? vv : 0.f;
    }
  }
}

// ---------------------------------------------------------------------------
// Dense bf16 GEMM: C(M,Nn) = A(M,Kp) * Wt(Nn,Kp)^T, fused bias/BN/ReLU.
// Block tile 16M x 128N (4 waves x 2 N-tiles), 64 K per barrier,
// 4 interleaved independent WMMAs per iteration; A-stream prefetch.
// grid = (ceil(M/16), Nn/128), block = 128.
// ---------------------------------------------------------------------------
__global__ __launch_bounds__(128)
void k_gemm(const unsigned short* __restrict__ A, const unsigned short* __restrict__ Wt,
            float* __restrict__ outF, unsigned short* __restrict__ outH,
            const float* __restrict__ bias,
            const float* __restrict__ bg, const float* __restrict__ bb,
            const float* __restrict__ bm, const float* __restrict__ bv,
            int M, int Nn, int Kp, int relu) {
  __shared__ __align__(16) unsigned short As[16][72];
  const int tid  = threadIdx.x;
  const int wave = tid >> 5, lane = tid & 31;
  const int m0 = blockIdx.x * 16;
  const int nA = blockIdx.y * 128 + wave * 16;
  const int nB = nA + 64;
  const int row = tid >> 3, seg = tid & 7;
  const unsigned short* arow = A + (size_t)(m0 + row) * Kp + seg * 8;
  const bool rv = (m0 + row) < M;
  v8f cA = {0.f, 0.f, 0.f, 0.f, 0.f, 0.f, 0.f, 0.f};
  v8f cB = {0.f, 0.f, 0.f, 0.f, 0.f, 0.f, 0.f, 0.f};
  const int bk = (lane >= 16) ? 8 : 0;
  const unsigned short* wrA = Wt + (size_t)(nA + (lane & 15)) * Kp;
  const unsigned short* wrB = Wt + (size_t)(nB + (lane & 15)) * Kp;
  for (int k0 = 0; k0 < Kp; k0 += 64) {
    u32x4 v = {0u, 0u, 0u, 0u};
    if (rv) {
      v = *(const u32x4*)(arow + k0);
      if (k0 + 64 < Kp) __builtin_prefetch(arow + k0 + 64, 0, 1);
    }
    *(u32x4*)&As[row][seg * 8] = v;
    __syncthreads();
    FragBF a0, a1, bA0, bA1, bB0, bB1;
    a0.q[0]  = *(const u32x4*)&As[lane & 15][bk];
    a0.q[1]  = *(const u32x4*)&As[lane & 15][16 + bk];
    a1.q[0]  = *(const u32x4*)&As[lane & 15][32 + bk];
    a1.q[1]  = *(const u32x4*)&As[lane & 15][48 + bk];
    bA0.q[0] = *(const u32x4*)(wrA + k0 + bk);
    bA0.q[1] = *(const u32x4*)(wrA + k0 + 16 + bk);
    bA1.q[0] = *(const u32x4*)(wrA + k0 + 32 + bk);
    bA1.q[1] = *(const u32x4*)(wrA + k0 + 48 + bk);
    bB0.q[0] = *(const u32x4*)(wrB + k0 + bk);
    bB0.q[1] = *(const u32x4*)(wrB + k0 + 16 + bk);
    bB1.q[0] = *(const u32x4*)(wrB + k0 + 32 + bk);
    bB1.q[1] = *(const u32x4*)(wrB + k0 + 48 + bk);
    cA = __builtin_amdgcn_wmma_f32_16x16x32_bf16(false, a0.v, false, bA0.v,
                                                 (short)0, cA, false, false);
    cB = __builtin_amdgcn_wmma_f32_16x16x32_bf16(false, a0.v, false, bB0.v,
                                                 (short)0, cB, false, false);
    cA = __builtin_amdgcn_wmma_f32_16x16x32_bf16(false, a1.v, false, bA1.v,
                                                 (short)0, cA, false, false);
    cB = __builtin_amdgcn_wmma_f32_16x16x32_bf16(false, a1.v, false, bB1.v,
                                                 (short)0, cB, false, false);
    __syncthreads();
  }
  const int hi = (lane >> 4) << 3;
#pragma unroll
  for (int tile = 0; tile < 2; ++tile) {
    const int n = (tile ? nB : nA) + (lane & 15);
    const v8f& c = tile ? cB : cA;
    float bs = bias ? bias[n] : 0.f;
    float sc = 1.f, sh = 0.f;
    if (bg) { sc = bg[n] * rsqrtf(bv[n] + 1e-5f); sh = bb[n] - bm[n] * sc; }
#pragma unroll
    for (int r = 0; r < 8; ++r) {
      int m = m0 + r + hi;
      if (m < M) {
        float vv = (c[r] + bs) * sc + sh;
        if (relu) vv = vv > 0.f ? vv : 0.f;
        if (outF) outF[(size_t)m * Nn + n] = vv;
        if (outH) outH[(size_t)m * Nn + n] = f2bf(vv);
      }
    }
  }
}

// ---------------------------------------------------------------------------
// 2x2x2 max-pool, stride 2, per-dim pad (implicit -inf padding).
// ---------------------------------------------------------------------------
__global__ void k_pool(const float* __restrict__ in, float* __restrict__ outF,
                       unsigned short* __restrict__ outH, int BC,
                       int D, int H, int W, int OD, int OH, int OW,
                       int pd, int ph, int pw) {
  int i = blockIdx.x * blockDim.x + threadIdx.x;
  if (i >= BC * OD * OH * OW) return;
  int ow = i % OW, t = i / OW;
  int oh = t % OH; t /= OH;
  int od = t % OD; t /= OD;
  const float* ib = in + (size_t)t * D * H * W;
  float mx = -3.4e38f;
  for (int dz = 0; dz < 2; ++dz) { int iz = 2 * od - pd + dz; if (iz < 0 || iz >= D) continue;
    for (int dy = 0; dy < 2; ++dy) { int iy = 2 * oh - ph + dy; if (iy < 0 || iy >= H) continue;
      for (int dx = 0; dx < 2; ++dx) { int ix = 2 * ow - pw + dx; if (ix < 0 || ix >= W) continue;
        float v = ib[((size_t)iz * H + iy) * W + ix]; mx = v > mx ? v : mx; } } }
  if (outF) outF[i] = mx;
  if (outH) outH[i] = f2bf(mx);
}

// ---------------------------------------------------------------------------
// GRU layer (H=12): one block per sequence, 25 sequential steps in LDS.
// ---------------------------------------------------------------------------
__global__ __launch_bounds__(64)
void k_gru(const float* __restrict__ xin, int In,
           const float* __restrict__ Wih, const float* __restrict__ Whh,
           const float* __restrict__ bih, const float* __restrict__ bhh,
           float* __restrict__ out) {
  __shared__ float h[GRUH], gi[36], gh[36], hn[GRUH];
  const int tid = threadIdx.x, seq = blockIdx.x;
  if (tid < GRUH) h[tid] = 0.f;
  __syncthreads();
  for (int t = 0; t < Tc; ++t) {
    const float* x = xin + (size_t)(seq * Tc + t) * In;
    if (tid < 36) {
      float a = bih[tid];
      const float* wr = Wih + (size_t)tid * In;
      for (int k = 0; k < In; ++k) a += wr[k] * x[k];
      gi[tid] = a;
      float b2 = bhh[tid];
      const float* wh = Whh + tid * GRUH;
      for (int k = 0; k < GRUH; ++k) b2 += wh[k] * h[k];
      gh[tid] = b2;
    }
    __syncthreads();
    if (tid < GRUH) {
      float r = 1.f / (1.f + expf(-(gi[tid] + gh[tid])));
      float z = 1.f / (1.f + expf(-(gi[12 + tid] + gh[12 + tid])));
      float n = tanhf(gi[24 + tid] + r * gh[24 + tid]);
      hn[tid] = (1.f - z) * n + z * h[tid];
    }
    __syncthreads();
    if (tid < GRUH) {
      h[tid] = hn[tid];
      out[(size_t)(seq * Tc + t) * GRUH + tid] = hn[tid];
    }
    __syncthreads();
  }
}

// ---------------------------------------------------------------------------
// Node features: xn = bng0(concat(node_x, enc[s0, low_res])) -> bf16, padded
// ---------------------------------------------------------------------------
__global__ void k_xn(const float* __restrict__ node_x, const int* __restrict__ low_res,
                     const float* __restrict__ enc,
                     const float* __restrict__ bg, const float* __restrict__ bb,
                     const float* __restrict__ bm, const float* __restrict__ bv,
                     unsigned short* __restrict__ xn) {
  int i = blockIdx.x * blockDim.x + threadIdx.x;
  if (i >= NTc * KP0) return;
  int node = i / KP0, f = i % KP0;
  unsigned short o = 0;
  if (f < NODEF) {
    int s0 = node / Nc;
    float val;
    if (f == 0) val = node_x[node];
    else {
      int lr = low_res[node];
      val = enc[(size_t)(s0 * S1c + lr) * (Tc * GRUH) + (f - 1)];
    }
    float sc = bg[f] * rsqrtf(bv[f] + 1e-5f);
    val = val * sc + (bb[f] - bm[f] * sc);
    o = f2bf(val);
  }
  xn[i] = o;
}

// ---------------------------------------------------------------------------
// Edge preprocessing: in-degree + mean edge-attr for self loops; combined list
// ---------------------------------------------------------------------------
__global__ void k_cnt(const int* __restrict__ eidx, const float* __restrict__ eattr,
                      float* __restrict__ cntr, float* __restrict__ la) {
  int i = blockIdx.x * blockDim.x + threadIdx.x;
  if (i >= REALc) return;
  int s0 = i / Ec, e = i % Ec;
  int d = eidx[(size_t)s0 * 2 * Ec + Ec + e] + s0 * Nc;
  atomicAdd(&cntr[d], 1.f);
  atomicAdd(&la[2 * d],     eattr[(size_t)i * 2]);
  atomicAdd(&la[2 * d + 1], eattr[(size_t)i * 2 + 1]);
}
__global__ void k_edges(const int* __restrict__ eidx, const float* __restrict__ eattr,
                        const float* __restrict__ cntr, const float* __restrict__ la,
                        int* __restrict__ srcA, int* __restrict__ dstA,
                        float* __restrict__ eaA, float* __restrict__ cntA) {
  int i = blockIdx.x * blockDim.x + threadIdx.x;
  if (i < REALc) {
    int s0 = i / Ec, e = i % Ec;
    srcA[i] = eidx[(size_t)s0 * 2 * Ec + e] + s0 * Nc;
    dstA[i] = eidx[(size_t)s0 * 2 * Ec + Ec + e] + s0 * Nc;
    eaA[2 * i]     = eattr[(size_t)i * 2];
    eaA[2 * i + 1] = eattr[(size_t)i * 2 + 1];
  } else if (i < ETc) {
    int n = i - REALc;
    srcA[i] = n; dstA[i] = n;
    float cd = fmaxf(cntr[n], 1.f);
    eaA[2 * i]     = la[2 * n] / cd;
    eaA[2 * i + 1] = la[2 * n + 1] / cd;
    cntA[n] = cntr[n] + 1.f;                 // +1 self loop
  }
}

// ---------------------------------------------------------------------------
// GATv2 edge kernels. Hh = 1<<shiftH heads, O = head dim (mult of 4).
// ---------------------------------------------------------------------------
__global__ void k_gat_logit4(const int* __restrict__ srcA, const int* __restrict__ dstA,
                             const float* __restrict__ eaA,
                             const float* __restrict__ xl, const float* __restrict__ xr,
                             const float* __restrict__ We, const float* __restrict__ att,
                             float* __restrict__ logit, unsigned* __restrict__ lmax,
                             int shiftH, int O) {
  const int Hh = 1 << shiftH;
  int i = blockIdx.x * blockDim.x + threadIdx.x;
  if (i >= ETc * Hh) return;
  int e = i >> shiftH, h = i & (Hh - 1);
  int s = srcA[e], d = dstA[e];
  int ho = Hh * O;
  float ea0 = eaA[2 * e], ea1 = eaA[2 * e + 1];
  const float4* pl = (const float4*)(xl + (size_t)s * ho + h * O);
  const float4* pr = (const float4*)(xr + (size_t)d * ho + h * O);
  const float4* w0 = (const float4*)(We + h * O);
  const float4* w1 = (const float4*)(We + ho + h * O);
  const float4* at = (const float4*)(att + h * O);
  float acc = 0.f;
  for (int k = 0; k < (O >> 2); ++k) {
    float4 l = pl[k], r = pr[k], a0 = w0[k], a1 = w1[k], aa = at[k];
    float z0 = l.x + r.x + ea0 * a0.x + ea1 * a1.x;
    float z1 = l.y + r.y + ea0 * a0.y + ea1 * a1.y;
    float z2 = l.z + r.z + ea0 * a0.z + ea1 * a1.z;
    float z3 = l.w + r.w + ea0 * a0.w + ea1 * a1.w;
    z0 = z0 > 0.f ? z0 : 0.2f * z0;
    z1 = z1 > 0.f ? z1 : 0.2f * z1;
    z2 = z2 > 0.f ? z2 : 0.2f * z2;
    z3 = z3 > 0.f ? z3 : 0.2f * z3;
    acc += aa.x * z0 + aa.y * z1 + aa.z * z2 + aa.w * z3;
  }
  logit[i] = acc;
  atomicMax(&lmax[(d << shiftH) + h], f2ord(acc));
}
// scalar logit (O==1) for gat3
__global__ void k_gat_logit1(const int* __restrict__ srcA, const int* __restrict__ dstA,
                             const float* __restrict__ eaA,
                             const float* __restrict__ xl, const float* __restrict__ xr,
                             const float* __restrict__ We, const float* __restrict__ att,
                             float* __restrict__ logit, unsigned* __restrict__ lmax) {
  int e = blockIdx.x * blockDim.x + threadIdx.x;
  if (e >= ETc) return;
  int s = srcA[e], d = dstA[e];
  float z = xl[s] + xr[d] + eaA[2 * e] * We[0] + eaA[2 * e + 1] * We[1];
  z = z > 0.f ? z : 0.2f * z;
  float acc = att[0] * z;
  logit[e] = acc;
  atomicMax(&lmax[d], f2ord(acc));
}
__global__ void k_gat_exp(const int* __restrict__ dstA, float* __restrict__ logit,
                          const unsigned* __restrict__ lmax, float* __restrict__ den,
                          int shiftH) {
  const int Hh = 1 << shiftH;
  int i = blockIdx.x * blockDim.x + threadIdx.x;
  if (i >= ETc * Hh) return;
  int e = i >> shiftH, h = i & (Hh - 1);
  int d = dstA[e];
  float ex = expf(logit[i] - ord2f(lmax[(d << shiftH) + h]));
  logit[i] = ex;
  atomicAdd(&den[(d << shiftH) + h], ex);
}
// one block per edge; blockDim = ho; edge scalars block-uniform
__global__ void k_gat_aggr_e(const int* __restrict__ srcA, const int* __restrict__ dstA,
                             const float* __restrict__ exv, const float* __restrict__ den,
                             const float* __restrict__ xl, float* __restrict__ num,
                             int shiftH, int shiftO) {
  __shared__ float alph[8];
  const int Hh = 1 << shiftH;
  const int e = blockIdx.x, c = threadIdx.x;
  const int ho = blockDim.x;
  const int s = srcA[e], d = dstA[e];
  if (c < Hh) alph[c] = exv[(e << shiftH) + c] / den[(d << shiftH) + c];
  __syncthreads();
  float alpha = alph[c >> shiftO];
  atomicAdd(&num[(size_t)d * ho + c], alpha * xl[(size_t)s * ho + c]);
}
// gat3 aggregation (ho == 1): one thread per edge
__global__ void k_gat3_aggr(const int* __restrict__ srcA, const int* __restrict__ dstA,
                            const float* __restrict__ exv, const float* __restrict__ den,
                            const float* __restrict__ xl, float* __restrict__ num) {
  int e = blockIdx.x * blockDim.x + threadIdx.x;
  if (e >= ETc) return;
  int s = srcA[e], d = dstA[e];
  atomicAdd(&num[d], (exv[e] / den[d]) * xl[s]);
}
// one block per node; blockDim = ho
__global__ void k_gat_final_n(const float* __restrict__ num, const float* __restrict__ cntA,
                              const float* __restrict__ bias,
                              const float* __restrict__ bg, const float* __restrict__ bb,
                              const float* __restrict__ bm, const float* __restrict__ bv,
                              float* __restrict__ outF, unsigned short* __restrict__ outH) {
  const int n = blockIdx.x, c = threadIdx.x;
  const int ho = blockDim.x;
  float v = num[(size_t)n * ho + c] / fmaxf(cntA[n], 1.f) + bias[c];
  float sc = bg[c] * rsqrtf(bv[c] + 1e-5f);
  v = v * sc + (bb[c] - bm[c] * sc);
  v = v > 0.f ? v : 0.f;
  if (outF) outF[(size_t)n * ho + c] = v;
  if (outH) outH[(size_t)n * ho + c] = f2bf(v);
}

// gat3 projections (ho = 1) and final sigmoid + mask
__global__ void k_gat3_proj(const float* __restrict__ h2,
                            const float* __restrict__ Wl, const float* __restrict__ Wr,
                            const float* __restrict__ bl, const float* __restrict__ br,
                            float* __restrict__ xl3, float* __restrict__ xr3) {
  int n = blockIdx.x * blockDim.x + threadIdx.x;
  if (n >= NTc) return;
  const float* r = h2 + (size_t)n * 128;
  float al = bl[0], ar = br[0];
  for (int k = 0; k < 128; ++k) { float x = r[k]; al += x * Wl[k]; ar += x * Wr[k]; }
  xl3[n] = al; xr3[n] = ar;
}
__global__ void k_final(const float* __restrict__ num, const float* __restrict__ cntA,
                        const float* __restrict__ bias,
                        const unsigned char* __restrict__ mask,
                        const float* __restrict__ yv, float* __restrict__ out) {
  int n = blockIdx.x * blockDim.x + threadIdx.x;
  if (n >= NTc) return;
  float v = num[n] / fmaxf(cntA[n], 1.f) + bias[0];
  float p = 1.f / (1.f + expf(-v));
  bool m = mask[n] != 0;
  out[n] = m ? p : 0.f;
  out[NTc + n] = m ? yv[n] : 0.f;
}

// ---------------------------------------------------------------------------
// Host orchestration
// ---------------------------------------------------------------------------
static inline int cdiv(long long a, long long b) { return (int)((a + b - 1) / b); }

// Input order: top-level insertion order, then params pytree with sorted dict
// keys (JAX convention). bn dicts: b,g,m,v. gat dicts: We,Wl,Wr,att,bias,bl,br.
// gru layer dicts: Whh,Wih,bhh,bih.
enum {
  I_X = 0, I_NODEX, I_LOWRES, I_EIDX, I_EATTR, I_MASK, I_Y,
  I_BN1 = 7, I_BN2 = 11, I_BN3 = 15, I_BNF1 = 19, I_BNF2 = 23,
  I_BNG0 = 27, I_BNG1 = 31, I_BNG2 = 35,
  I_C1B = 39, I_C1W = 40, I_C2B = 41, I_C2W = 42, I_C3B = 43, I_C3W = 44,
  I_FC1B = 45, I_FC1W = 46, I_FC2B = 47, I_FC2W = 48,
  I_GAT1 = 49, I_GAT2 = 56, I_GAT3 = 63,
  I_GRU0 = 70, I_GRU1 = 74
};

extern "C" void kernel_launch(void* const* d_in, const int* in_sizes, int n_in,
                              void* d_out, int out_size, void* d_ws, size_t ws_size,
                              hipStream_t stream) {
  (void)in_sizes; (void)n_in; (void)out_size; (void)ws_size;
  auto F = [&](int i) { return (const float*)d_in[i]; };
  const float* X        = F(I_X);
  const float* node_x   = F(I_NODEX);
  const int*   low_res  = (const int*)d_in[I_LOWRES];
  const int*   eidx     = (const int*)d_in[I_EIDX];
  const float* eattr    = F(I_EATTR);
  const unsigned char* mask = (const unsigned char*)d_in[I_MASK];
  const float* yv       = F(I_Y);

  // workspace bump allocator (256B aligned)
  size_t off = 0;
  auto alloc = [&](size_t bytes) -> void* {
    void* p = (char*)d_ws + off;
    off += (bytes + 255) & ~(size_t)255;
    return p;
  };
  typedef unsigned short us;
  us* Wc1  = (us*)alloc((size_t)64 * 160 * 2);
  us* Wc2  = (us*)alloc((size_t)64 * 1728 * 2);
  us* Wc3  = (us*)alloc((size_t)256 * 1728 * 2);
  us* Wfc1 = (us*)alloc((size_t)512 * 2048 * 2);
  us* Wfc2 = (us*)alloc((size_t)256 * 512 * 2);
  us* Wl1  = (us*)alloc((size_t)256 * KP0 * 2);
  us* Wr1  = (us*)alloc((size_t)256 * KP0 * 2);
  us* Wl2  = (us*)alloc((size_t)128 * 256 * 2);
  us* Wr2  = (us*)alloc((size_t)128 * 256 * 2);
  int* TAB1 = (int*)alloc((size_t)192 * 160 * 4);          // conv1 im2col table
  // regionA: conv1 out (3200*64*180) then conv3 out (3200*256*48) — reuse
  float* regionA = (float*)alloc((size_t)Bc * 256 * 48 * 4);
  float* A1 = regionA;                   // (B,64,180)
  float* A3 = regionA;                   // (B,256,48)
  // regionB: conv2 out; after pool1 reused for GAT xl/xr/num
  float* regionB = (float*)alloc((size_t)Bc * 64 * 180 * 4);
  float* A2  = regionB;
  float* XL  = regionB;                                    // 40000*256
  float* XR  = regionB + (size_t)NTc * 256;
  float* NUM = regionB + (size_t)NTc * 512;
  float* P1   = (float*)alloc((size_t)Bc * 64 * 48 * 4);   // pool1 out
  us*    P2h  = (us*)alloc((size_t)Bc * 2048 * 2);         // pool2 out (bf16)
  us*    FC1h = (us*)alloc((size_t)Bc * 512 * 2);
  float* FC2f = (float*)alloc((size_t)Bc * 256 * 4);
  float* G1   = (float*)alloc((size_t)128 * Tc * GRUH * 4);
  float* ENC  = (float*)alloc((size_t)128 * Tc * GRUH * 4);
  us*    XN   = (us*)alloc((size_t)NTc * KP0 * 2);
  us*    H2h  = (us*)alloc((size_t)NTc * 256 * 2);         // gat1 out (bf16)
  float* H3f  = (float*)alloc((size_t)NTc * 128 * 4);      // gat2 out (f32)
  float* XL3  = (float*)alloc((size_t)NTc * 4);
  float* XR3  = (float*)alloc((size_t)NTc * 4);
  float* LOGIT = (float*)alloc((size_t)ETc * 2 * 4);
  unsigned* LMAX = (unsigned*)alloc((size_t)NTc * 2 * 4);
  float* DEN  = (float*)alloc((size_t)NTc * 2 * 4);
  float* CNTR = (float*)alloc((size_t)NTc * 4);
  float* CNTA = (float*)alloc((size_t)NTc * 4);
  float* LA   = (float*)alloc((size_t)NTc * 2 * 4);
  int*   SRC  = (int*)alloc((size_t)ETc * 4);
  int*   DST  = (int*)alloc((size_t)ETc * 4);
  float* EA   = (float*)alloc((size_t)ETc * 2 * 4);

  const int TB = 256;
  // ---- weight prep (bf16, padded/transposed; convs 2&3 tap-major) ----
  k_wpad<<<cdiv(64 * 160, TB), TB, 0, stream>>>(F(I_C1W), Wc1, 64, 135, 160);
  k_wpad_tap<<<cdiv(64 * 1728, TB), TB, 0, stream>>>(F(I_C2W), Wc2, 64, 64);
  k_wpad_tap<<<cdiv(256 * 1728, TB), TB, 0, stream>>>(F(I_C3W), Wc3, 256, 64);
  k_wtrans<<<cdiv(512 * 2048, TB), TB, 0, stream>>>(F(I_FC1W), Wfc1, 512, 2048, 2048);
  k_wtrans<<<cdiv(256 * 512, TB), TB, 0, stream>>>(F(I_FC2W), Wfc2, 256, 512, 512);
  k_wtrans<<<cdiv(256 * KP0, TB), TB, 0, stream>>>(F(I_GAT1 + 1), Wl1, 256, NODEF, KP0);
  k_wtrans<<<cdiv(256 * KP0, TB), TB, 0, stream>>>(F(I_GAT1 + 2), Wr1, 256, NODEF, KP0);
  k_wtrans<<<cdiv(128 * 256, TB), TB, 0, stream>>>(F(I_GAT2 + 1), Wl2, 128, 256, 256);
  k_wtrans<<<cdiv(128 * 256, TB), TB, 0, stream>>>(F(I_GAT2 + 2), Wr2, 128, 256, 256);
  k_tab1<<<cdiv(192 * 160, TB), TB, 0, stream>>>(TAB1);

  // ---- CNN (implicit-GEMM WMMA convs + pools) ----
  k_conv1_tab<<<dim3(12, 1, Bc), 128, 0, stream>>>(
      X, Wc1, A1, F(I_C1B), F(I_BN1 + 1), F(I_BN1 + 0), F(I_BN1 + 2), F(I_BN1 + 3), TAB1);
  k_conv_tap<<<dim3(12, 1, Bc), 128, 0, stream>>>(
      A1, Wc2, A2, F(I_C2B), F(I_BN2 + 1), F(I_BN2 + 0), F(I_BN2 + 2), F(I_BN2 + 3),
      5, 6, 6, 64);
  k_pool<<<cdiv((size_t)Bc * 64 * 48, TB), TB, 0, stream>>>(
      A2, P1, nullptr, Bc * 64, 5, 6, 6, 3, 4, 4, 1, 1, 1);
  k_conv_tap<<<dim3(3, 4, Bc), 128, 0, stream>>>(
      P1, Wc3, A3, F(I_C3B), F(I_BN3 + 1), F(I_BN3 + 0), F(I_BN3 + 2), F(I_BN3 + 3),
      3, 4, 4, 256);
  k_pool<<<cdiv((size_t)Bc * 256 * 8, TB), TB, 0, stream>>>(
      A3, nullptr, P2h, Bc * 256, 3, 4, 4, 2, 2, 2, 1, 0, 0);

  // ---- FC layers (WMMA GEMM) ----
  k_gemm<<<dim3(200, 4), 128, 0, stream>>>(
      P2h, Wfc1, nullptr, FC1h, F(I_FC1B),
      F(I_BNF1 + 1), F(I_BNF1 + 0), F(I_BNF1 + 2), F(I_BNF1 + 3), Bc, 512, 2048, 1);
  k_gemm<<<dim3(200, 2), 128, 0, stream>>>(
      FC1h, Wfc2, FC2f, nullptr, F(I_FC2B),
      F(I_BNF2 + 1), F(I_BNF2 + 0), F(I_BNF2 + 2), F(I_BNF2 + 3), Bc, 256, 512, 1);

  // ---- GRU (2 layers) ----
  k_gru<<<128, 64, 0, stream>>>(FC2f, 256, F(I_GRU0 + 1), F(I_GRU0 + 0),
                                F(I_GRU0 + 3), F(I_GRU0 + 2), G1);
  k_gru<<<128, 64, 0, stream>>>(G1, GRUH, F(I_GRU1 + 1), F(I_GRU1 + 0),
                                F(I_GRU1 + 3), F(I_GRU1 + 2), ENC);

  // ---- graph preprocessing ----
  hipMemsetAsync(CNTR, 0, (size_t)NTc * 4, stream);
  hipMemsetAsync(LA, 0, (size_t)NTc * 8, stream);
  k_cnt<<<cdiv(REALc, TB), TB, 0, stream>>>(eidx, eattr, CNTR, LA);
  k_edges<<<cdiv(ETc, TB), TB, 0, stream>>>(eidx, eattr, CNTR, LA, SRC, DST, EA, CNTA);
  k_xn<<<cdiv((size_t)NTc * KP0, TB), TB, 0, stream>>>(
      node_x, low_res, ENC, F(I_BNG0 + 1), F(I_BNG0 + 0), F(I_BNG0 + 2), F(I_BNG0 + 3), XN);

  // ---- GAT layer 1 (heads=2, out=128) ----
  k_gemm<<<dim3(2500, 2), 128, 0, stream>>>(XN, Wl1, XL, nullptr, F(I_GAT1 + 5),
      nullptr, nullptr, nullptr, nullptr, NTc, 256, KP0, 0);
  k_gemm<<<dim3(2500, 2), 128, 0, stream>>>(XN, Wr1, XR, nullptr, F(I_GAT1 + 6),
      nullptr, nullptr, nullptr, nullptr, NTc, 256, KP0, 0);
  hipMemsetAsync(LMAX, 0, (size_t)NTc * 2 * 4, stream);
  hipMemsetAsync(DEN, 0, (size_t)NTc * 2 * 4, stream);
  hipMemsetAsync(NUM, 0, (size_t)NTc * 256 * 4, stream);
  k_gat_logit4<<<cdiv(ETc * 2, TB), TB, 0, stream>>>(SRC, DST, EA, XL, XR,
      F(I_GAT1 + 0), F(I_GAT1 + 3), LOGIT, LMAX, 1, 128);
  k_gat_exp<<<cdiv(ETc * 2, TB), TB, 0, stream>>>(DST, LOGIT, LMAX, DEN, 1);
  k_gat_aggr_e<<<ETc, 256, 0, stream>>>(SRC, DST, LOGIT, DEN, XL, NUM, 1, 7);
  k_gat_final_n<<<NTc, 256, 0, stream>>>(NUM, CNTA, F(I_GAT1 + 4),
      F(I_BNG1 + 1), F(I_BNG1 + 0), F(I_BNG1 + 2), F(I_BNG1 + 3), nullptr, H2h);

  // ---- GAT layer 2 (heads=1, out=128) ----
  k_gemm<<<dim3(2500, 1), 128, 0, stream>>>(H2h, Wl2, XL, nullptr, F(I_GAT2 + 5),
      nullptr, nullptr, nullptr, nullptr, NTc, 128, 256, 0);
  k_gemm<<<dim3(2500, 1), 128, 0, stream>>>(H2h, Wr2, XR, nullptr, F(I_GAT2 + 6),
      nullptr, nullptr, nullptr, nullptr, NTc, 128, 256, 0);
  hipMemsetAsync(LMAX, 0, (size_t)NTc * 4, stream);
  hipMemsetAsync(DEN, 0, (size_t)NTc * 4, stream);
  hipMemsetAsync(NUM, 0, (size_t)NTc * 128 * 4, stream);
  k_gat_logit4<<<cdiv(ETc, TB), TB, 0, stream>>>(SRC, DST, EA, XL, XR,
      F(I_GAT2 + 0), F(I_GAT2 + 3), LOGIT, LMAX, 0, 128);
  k_gat_exp<<<cdiv(ETc, TB), TB, 0, stream>>>(DST, LOGIT, LMAX, DEN, 0);
  k_gat_aggr_e<<<ETc, 128, 0, stream>>>(SRC, DST, LOGIT, DEN, XL, NUM, 0, 7);
  k_gat_final_n<<<NTc, 128, 0, stream>>>(NUM, CNTA, F(I_GAT2 + 4),
      F(I_BNG2 + 1), F(I_BNG2 + 0), F(I_BNG2 + 2), F(I_BNG2 + 3), H3f, nullptr);

  // ---- GAT layer 3 (heads=1, out=1) + sigmoid + mask ----
  k_gat3_proj<<<cdiv(NTc, TB), TB, 0, stream>>>(H3f, F(I_GAT3 + 1), F(I_GAT3 + 2),
      F(I_GAT3 + 5), F(I_GAT3 + 6), XL3, XR3);
  hipMemsetAsync(LMAX, 0, (size_t)NTc * 4, stream);
  hipMemsetAsync(DEN, 0, (size_t)NTc * 4, stream);
  hipMemsetAsync(NUM, 0, (size_t)NTc * 4, stream);
  k_gat_logit1<<<cdiv(ETc, TB), TB, 0, stream>>>(SRC, DST, EA, XL3, XR3,
      F(I_GAT3 + 0), F(I_GAT3 + 3), LOGIT, LMAX);
  k_gat_exp<<<cdiv(ETc, TB), TB, 0, stream>>>(DST, LOGIT, LMAX, DEN, 0);
  k_gat3_aggr<<<cdiv(ETc, TB), TB, 0, stream>>>(SRC, DST, LOGIT, DEN, XL3, NUM);
  k_final<<<cdiv(NTc, TB), TB, 0, stream>>>(NUM, CNTA, F(I_GAT3 + 4), mask, yv, (float*)d_out);
}
